// DecoderStepCached_59184649339020
// MI455X (gfx1250) — matrix operand order repaired and checked by reading, not compiled, not run on hardware
//
#include <hip/hip_runtime.h>
#include <hip/hip_bf16.h>

// ---------------- constants from the reference ----------------
#define Lyr    12
#define NH     16
#define HD     64
#define DM     1024
#define DFF    4096
#define MAXLEN 1024
#define TENC   1024
#define VOCAB  32000
#define SCALE  0.125f      // 1/sqrt(64)
#define NEG_BIG (-3.0e38f)

typedef __attribute__((ext_vector_type(16))) __bf16 v16bf;
typedef __attribute__((ext_vector_type(8)))  __bf16 v8bf;
typedef __attribute__((ext_vector_type(4)))  __bf16 v4bf;
typedef __attribute__((ext_vector_type(2)))  __bf16 v2bf;
typedef __attribute__((ext_vector_type(8)))  float  v8f;

// ================= embedding: hidden = tok_emb[id] + pos_emb[step] =========
__global__ __launch_bounds__(256)
void embed_kernel(const int* __restrict__ input_id, const int* __restrict__ step,
                  const float* __restrict__ tok_emb, const float* __restrict__ pos_emb,
                  float* __restrict__ hidden) {
    int d = blockIdx.x * 256 + threadIdx.x;      // grid 4
    int id = input_id[0];
    int s  = step[0];
    hidden[d] = tok_emb[(size_t)id * DM + d] + pos_emb[(size_t)s * DM + d];
}

// ================= fp32 -> bf16 bulk convert (for the reused A matrix) =====
__global__ __launch_bounds__(256)
void f32_to_bf16_kernel(const float* __restrict__ in, __bf16* __restrict__ out) {
    int i = (blockIdx.x * 256 + threadIdx.x) * 4;
    float4 v = *(const float4*)(in + i);
    v4bf o;
    o[0] = (__bf16)v.x; o[1] = (__bf16)v.y; o[2] = (__bf16)v.z; o[3] = (__bf16)v.w;
    *(v4bf*)(out + i) = o;
}

// ================= layer norm over D=1024 (one block, 256 threads) =========
__global__ __launch_bounds__(256)
void ln_kernel(const float* __restrict__ x, const float* __restrict__ sc,
               const float* __restrict__ bi, float* __restrict__ y) {
    __shared__ float red[256];
    int tid = threadIdx.x;
    float v0 = x[tid], v1 = x[tid + 256], v2 = x[tid + 512], v3 = x[tid + 768];
    red[tid] = v0 + v1 + v2 + v3;
    __syncthreads();
    for (int o = 128; o > 0; o >>= 1) { if (tid < o) red[tid] += red[tid + o]; __syncthreads(); }
    float mean = red[0] * (1.0f / DM);
    __syncthreads();
    float d0 = v0 - mean, d1 = v1 - mean, d2 = v2 - mean, d3 = v3 - mean;
    red[tid] = d0 * d0 + d1 * d1 + d2 * d2 + d3 * d3;
    __syncthreads();
    for (int o = 128; o > 0; o >>= 1) { if (tid < o) red[tid] += red[tid + o]; __syncthreads(); }
    float rstd = rsqrtf(red[0] * (1.0f / DM) + 1e-5f);
    y[tid]       = d0 * rstd * sc[tid]       + bi[tid];
    y[tid + 256] = d1 * rstd * sc[tid + 256] + bi[tid + 256];
    y[tid + 512] = d2 * rstd * sc[tid + 512] + bi[tid + 512];
    y[tid + 768] = d3 * rstd * sc[tid + 768] + bi[tid + 768];
}

// ================= matvec: out[j] (=|+=) act(sum_d x[d]*W[d,j] + b[j]) =====
__global__ __launch_bounds__(256)
void matvec_kernel(const float* __restrict__ x, const float* __restrict__ W,
                   const float* __restrict__ bias, float* __restrict__ out,
                   int K, int N, int relu, int accum) {
    __shared__ float xs[4096];
    int tid = threadIdx.x;
    for (int i = tid; i < K; i += 256) xs[i] = x[i];
    __syncthreads();
    int j = blockIdx.x * 256 + tid;
    if (j >= N) return;
    float acc = bias ? bias[j] : 0.0f;
    const float* Wj = W + j;
    for (int d = 0; d < K; d += 4) {
        acc += xs[d]     * Wj[(size_t)d * N];
        acc += xs[d + 1] * Wj[(size_t)(d + 1) * N];
        acc += xs[d + 2] * Wj[(size_t)(d + 2) * N];
        acc += xs[d + 3] * Wj[(size_t)(d + 3) * N];
    }
    if (relu) acc = fmaxf(acc, 0.0f);
    if (accum) out[j] += acc; else out[j] = acc;
}

// ================= write new k/v into cache slot `step` =====================
__global__ __launch_bounds__(256)
void kvwrite_kernel(const float* __restrict__ kn, const float* __restrict__ vn,
                    const int* __restrict__ step,
                    float* __restrict__ ck, float* __restrict__ cv) {
    int d = blockIdx.x * 256 + threadIdx.x;      // grid 4 -> d in [0,1024)
    int s = step[0];
    int h = d >> 6, hd = d & 63;
    size_t off = ((size_t)h * MAXLEN + s) * HD + hd;
    ck[off] = kn[d];
    cv[off] = vn[d];
}

// ================= attention (one head per block) ===========================
__global__ __launch_bounds__(256)
void attn_kernel(const float* __restrict__ q, const float* __restrict__ Kc,
                 const float* __restrict__ Vc, const float* __restrict__ amask,
                 const int* __restrict__ step, int len, long sH, long sT,
                 int useStep, float* __restrict__ out) {
    __shared__ float scb[1024];
    __shared__ float red[256];
    __shared__ float qh[64];
    const int h = blockIdx.x, tid = threadIdx.x;
    if (tid < 64) qh[tid] = q[h * 64 + tid];
    __syncthreads();
    const int Tv = useStep ? (step[0] + 1) : len;
    float lmax = NEG_BIG;
    for (int t = tid; t < len; t += 256) {
        float v;
        if (t < Tv) {
            const float4* kp = (const float4*)(Kc + (size_t)h * sH + (size_t)t * sT);
            const float4* qp = (const float4*)qh;
            float d = 0.0f;
            #pragma unroll
            for (int i = 0; i < 16; ++i) {
                float4 a = qp[i], c = kp[i];
                d += a.x * c.x + a.y * c.y + a.z * c.z + a.w * c.w;
            }
            v = d * SCALE + (amask ? amask[t] : 0.0f);
        } else {
            v = NEG_BIG;
        }
        scb[t] = v;
        lmax = fmaxf(lmax, v);
    }
    red[tid] = lmax; __syncthreads();
    for (int o = 128; o > 0; o >>= 1) { if (tid < o) red[tid] = fmaxf(red[tid], red[tid + o]); __syncthreads(); }
    const float mx = red[0];
    __syncthreads();
    float lsum = 0.0f;
    for (int t = tid; t < len; t += 256) { float p = __expf(scb[t] - mx); scb[t] = p; lsum += p; }
    red[tid] = lsum; __syncthreads();
    for (int o = 128; o > 0; o >>= 1) { if (tid < o) red[tid] += red[tid + o]; __syncthreads(); }
    const float ssum = red[0];
    __syncthreads();
    const int hd = tid & 63, g = tid >> 6;
    float acc = 0.0f;
    for (int t = g; t < Tv; t += 4)
        acc += scb[t] * Vc[(size_t)h * sH + (size_t)t * sT + hd];
    red[tid] = acc; __syncthreads();
    if (tid < 64)
        out[h * 64 + tid] = (red[tid] + red[tid + 64] + red[tid + 128] + red[tid + 192]) / ssum;
}

// ================= WMMA GEMM: C[M,N] = A[M,K] @ B[K,N] + bias[N] ===========
// A: pre-converted bf16 [M,K] row-major.  B: fp32 [K,N], converted inline.
// Block = 256 thr (8 waves); block tile 64(M) x 128(N); waves 2x4; each wave
// owns a 32x32 macro-tile (4 accumulators) -> 4 v_wmma per K-step with
// 2 A-frags + 2 B-frags (each fragment reused twice).
#define LDK 40   // LDS row stride in bf16 elems (80B: 16B-multiple + conflict pad)
__global__ __launch_bounds__(256)
void gemm_bf16_wmma_kernel(const __bf16* __restrict__ A, const float* __restrict__ B,
                           const float* __restrict__ bias, float* __restrict__ C,
                           int M, int N, int K) {
    __shared__ __align__(16) __bf16 Abf[64][LDK];    // [m][k]
    __shared__ __align__(16) __bf16 Bt [128][LDK];   // [n][k] (transposed)
    const int tid  = threadIdx.x;
    const int wid  = tid >> 5, lane = tid & 31;
    const int m0   = blockIdx.y * 64;
    const int n0   = blockIdx.x * 128;
    const int wm2  = (wid >> 2) * 32;   // 0 / 32
    const int wn2  = (wid & 3) * 32;    // 0..96
    const int lr   = lane & 15;
    const int hi   = lane >> 4;         // half-wave select

    v8f acc[2][2] = {};
    for (int k0 = 0; k0 < K; k0 += 32) {
        // stage A tile 64x32 bf16: one b128 global load + one b128 LDS store/thread
        {
            int idx = tid * 8;                    // 2048 elems
            int m = idx >> 5, kk = idx & 31;
            v8bf av = *(const v8bf*)(A + (size_t)(m0 + m) * K + (k0 + kk));
            *(v8bf*)(&Abf[m][kk]) = av;
        }
        // stage B tile 32x128 fp32 -> bf16 transposed.  Each group handles a
        // (k,k+1) row pair over 4 columns: 2 coalesced float4 loads, then pack
        // k-adjacent pairs into aligned 32-bit LDS stores (4 x ds_store_b32).
        #pragma unroll
        for (int i = 0; i < 2; ++i) {
            int g = tid + i * 256;                // 512 groups
            int k = (g >> 5) * 2;                 // even k
            int n4 = (g & 31) * 4;
            float4 va = *(const float4*)(B + (size_t)(k0 + k)     * N + (n0 + n4));
            float4 vb = *(const float4*)(B + (size_t)(k0 + k + 1) * N + (n0 + n4));
            v2bf p0, p1, p2, p3;
            p0[0] = (__bf16)va.x; p0[1] = (__bf16)vb.x;
            p1[0] = (__bf16)va.y; p1[1] = (__bf16)vb.y;
            p2[0] = (__bf16)va.z; p2[1] = (__bf16)vb.z;
            p3[0] = (__bf16)va.w; p3[1] = (__bf16)vb.w;
            *(v2bf*)(&Bt[n4 + 0][k]) = p0;
            *(v2bf*)(&Bt[n4 + 1][k]) = p1;
            *(v2bf*)(&Bt[n4 + 2][k]) = p2;
            *(v2bf*)(&Bt[n4 + 3][k]) = p3;
        }
        __syncthreads();
        // A fragments: lanes 0-15 -> K {0..7,16..23}; lanes 16-31 -> K {8..15,24..31}
        v16bf af[2];
        #pragma unroll
        for (int mi = 0; mi < 2; ++mi) {
            const __bf16* arow = &Abf[wm2 + mi * 16 + lr][0];
            v8bf alo = *(const v8bf*)(arow + hi * 8);
            v8bf ahi = *(const v8bf*)(arow + hi * 8 + 16);
            af[mi] = __builtin_shufflevector(alo, ahi,
                0, 1, 2, 3, 4, 5, 6, 7, 8, 9, 10, 11, 12, 13, 14, 15);
        }
        // B fragments: lane = column; lanes 0-15 -> K {0..7,16..23}; lanes 16-31 -> K {8..15,24..31}
        v16bf bfr[2];
        #pragma unroll
        for (int ni = 0; ni < 2; ++ni) {
            const __bf16* brow = &Bt[wn2 + ni * 16 + lr][0];
            v8bf blo = *(const v8bf*)(brow + hi * 8);
            v8bf bhi = *(const v8bf*)(brow + hi * 8 + 16);
            bfr[ni] = __builtin_shufflevector(blo, bhi,
                0, 1, 2, 3, 4, 5, 6, 7, 8, 9, 10, 11, 12, 13, 14, 15);
        }
        #pragma unroll
        for (int mi = 0; mi < 2; ++mi)
            #pragma unroll
            for (int ni = 0; ni < 2; ++ni)
                acc[mi][ni] = __builtin_amdgcn_wmma_f32_16x16x32_bf16(
                    false, af[mi], false, bfr[ni], (short)0, acc[mi][ni], false, false);
        __syncthreads();
    }
    // D layout: VGPR r -> row r (lanes 0-15) / r+8 (lanes 16-31); col = lane%16
    #pragma unroll
    for (int mi = 0; mi < 2; ++mi) {
        #pragma unroll
        for (int ni = 0; ni < 2; ++ni) {
            const int col = n0 + wn2 + ni * 16 + lr;
            const float bvv = bias ? bias[col] : 0.0f;
            #pragma unroll
            for (int r = 0; r < 8; ++r) {
                int row = m0 + wm2 + mi * 16 + r + hi * 8;
                C[(size_t)row * N + col] = acc[mi][ni][r] + bvv;
            }
        }
    }
}

// ============================ host-side launch =============================
extern "C" void kernel_launch(void* const* d_in, const int* in_sizes, int n_in,
                              void* d_out, int out_size, void* d_ws, size_t ws_size,
                              hipStream_t stream) {
    const float* enc      = (const float*)d_in[0];
    const int*   input_id = (const int*)  d_in[1];
    const float* cache_k  = (const float*)d_in[2];
    const float* cache_v  = (const float*)d_in[3];
    const int*   step     = (const int*)  d_in[4];
    const float* cmask    = (const float*)d_in[5];
    const float* tok_emb  = (const float*)d_in[6];
    const float* pos_emb  = (const float*)d_in[7];
    const float* ln1_s = (const float*)d_in[8],  *ln1_b = (const float*)d_in[9];
    const float* ln2_s = (const float*)d_in[10], *ln2_b = (const float*)d_in[11];
    const float* ln3_s = (const float*)d_in[12], *ln3_b = (const float*)d_in[13];
    const float* lnf_s = (const float*)d_in[14], *lnf_b = (const float*)d_in[15];
    const float* Wq  = (const float*)d_in[16], *bq  = (const float*)d_in[17];
    const float* Wk  = (const float*)d_in[18], *bk  = (const float*)d_in[19];
    const float* Wv  = (const float*)d_in[20], *bv  = (const float*)d_in[21];
    const float* Wo  = (const float*)d_in[22], *bo  = (const float*)d_in[23];
    const float* Wq2 = (const float*)d_in[24], *bq2 = (const float*)d_in[25];
    const float* Wk2 = (const float*)d_in[26], *bk2 = (const float*)d_in[27];
    const float* Wv2 = (const float*)d_in[28], *bv2 = (const float*)d_in[29];
    const float* Wo2 = (const float*)d_in[30], *bo2 = (const float*)d_in[31];
    const float* W1  = (const float*)d_in[32], *b1  = (const float*)d_in[33];
    const float* W2  = (const float*)d_in[34], *b2  = (const float*)d_in[35];
    const float* Wc  = (const float*)d_in[36], *bc  = (const float*)d_in[37];

    const size_t CACHE_ELEMS = (size_t)Lyr * NH * MAXLEN * HD;   // 12,582,912
    float* logits = (float*)d_out;
    float* ck_out = logits + VOCAB;
    float* cv_out = ck_out + CACHE_ELEMS;

    // workspace layout (floats)
    float* ws     = (float*)d_ws;
    float* hidden = ws;                  // 1024
    float* xln    = ws + 1024;           // 1024
    float* qb     = ws + 2048;           // 1024
    float* knb    = ws + 3072;           // 1024
    float* vnb    = ws + 4096;           // 1024
    float* attnb  = ws + 5120;           // 1024
    float* ffb    = ws + 6144;           // 4096
    float* q2b    = ws + 10240;          // 1024
    float* kcb    = ws + 16384;          // 1024*1024
    float* vcb    = kcb + (size_t)TENC * DM;
    __bf16* encbf = (__bf16*)(vcb + (size_t)TENC * DM);  // 1024*1024 bf16 (2MB)

    // 1) pass-through caches into outputs (slots overwritten per layer below)
    hipMemcpyAsync(ck_out, cache_k, CACHE_ELEMS * sizeof(float),
                   hipMemcpyDeviceToDevice, stream);
    hipMemcpyAsync(cv_out, cache_v, CACHE_ELEMS * sizeof(float),
                   hipMemcpyDeviceToDevice, stream);

    // 2) embedding + one-time bf16 conversion of encoder states (reused 24x)
    embed_kernel<<<4, 256, 0, stream>>>(input_id, step, tok_emb, pos_emb, hidden);
    f32_to_bf16_kernel<<<(TENC * DM) / 1024, 256, 0, stream>>>(enc, encbf);

    const dim3 gemm_grid(DM / 128, TENC / 64);   // (8, 16)

    for (int i = 0; i < Lyr; ++i) {
        const size_t Woff = (size_t)i * DM * DM;
        const size_t boff = (size_t)i * DM;
        float* ckL = ck_out + (size_t)i * NH * MAXLEN * HD;
        float* cvL = cv_out + (size_t)i * NH * MAXLEN * HD;

        // ---- self attention ----
        ln_kernel<<<1, 256, 0, stream>>>(hidden, ln1_s + boff, ln1_b + boff, xln);
        matvec_kernel<<<4, 256, 0, stream>>>(xln, Wq + Woff, bq + boff, qb,  DM, DM, 0, 0);
        matvec_kernel<<<4, 256, 0, stream>>>(xln, Wk + Woff, bk + boff, knb, DM, DM, 0, 0);
        matvec_kernel<<<4, 256, 0, stream>>>(xln, Wv + Woff, bv + boff, vnb, DM, DM, 0, 0);
        kvwrite_kernel<<<4, 256, 0, stream>>>(knb, vnb, step, ckL, cvL);
        attn_kernel<<<NH, 256, 0, stream>>>(qb, ckL, cvL, nullptr, step, MAXLEN,
                                            (long)MAXLEN * HD, (long)HD, 1, attnb);
        matvec_kernel<<<4, 256, 0, stream>>>(attnb, Wo + Woff, bo + boff, hidden, DM, DM, 0, 1);

        // ---- cross attention ----
        ln_kernel<<<1, 256, 0, stream>>>(hidden, ln2_s + boff, ln2_b + boff, xln);
        matvec_kernel<<<4, 256, 0, stream>>>(xln, Wq2 + Woff, bq2 + boff, q2b, DM, DM, 0, 0);
        gemm_bf16_wmma_kernel<<<gemm_grid, 256, 0, stream>>>(
            encbf, Wk2 + Woff, bk2 + boff, kcb, TENC, DM, DM);
        gemm_bf16_wmma_kernel<<<gemm_grid, 256, 0, stream>>>(
            encbf, Wv2 + Woff, bv2 + boff, vcb, TENC, DM, DM);
        attn_kernel<<<NH, 256, 0, stream>>>(q2b, kcb, vcb, cmask, step, TENC,
                                            (long)HD, (long)DM, 0, attnb);
        matvec_kernel<<<4, 256, 0, stream>>>(attnb, Wo2 + Woff, bo2 + boff, hidden, DM, DM, 0, 1);

        // ---- FFN ----
        ln_kernel<<<1, 256, 0, stream>>>(hidden, ln3_s + boff, ln3_b + boff, xln);
        matvec_kernel<<<DFF / 256, 256, 0, stream>>>(
            xln, W1 + (size_t)i * DM * DFF, b1 + (size_t)i * DFF, ffb, DM, DFF, 1, 0);
        matvec_kernel<<<4, 256, 0, stream>>>(
            ffb, W2 + (size_t)i * DFF * DM, b2 + boff, hidden, DFF, DM, 0, 1);
    }

    // 3) final LN + vocab head
    ln_kernel<<<1, 256, 0, stream>>>(hidden, lnf_s, lnf_b, xln);
    matvec_kernel<<<(VOCAB + 255) / 256, 256, 0, stream>>>(
        xln, Wc, bc, logits, DM, VOCAB, 0, 0);
}
